// TorchMDNETEncoder_26079041421824
// MI455X (gfx1250) — compile-verified, matrix operand order checked by reading, not run on hardware
//
#include <hip/hip_runtime.h>
#include <hip/hip_bf16.h>

typedef __attribute__((ext_vector_type(16))) _Float16 v16h;
typedef __attribute__((ext_vector_type(8)))  _Float16 v8h;
typedef __attribute__((ext_vector_type(8)))  float    v8f;
typedef __attribute__((ext_vector_type(4)))  unsigned int u32x4;
typedef __attribute__((ext_vector_type(8)))  int i32x8;
typedef __attribute__((ext_vector_type(4)))  int i32x4;

#define HIDC 128
#define NLAY 3
#define RBFC 64
#define CUTV 5.0f

__device__ __forceinline__ float siluf(float x){ return x / (1.f + __expf(-x)); }

__device__ __forceinline__ v8f wmma_f16(v16h a, v16h b, v8f c){
  return __builtin_amdgcn_wmma_f32_16x16x32_f16(false, a, false, b, (short)0, c, false, false);
}

// A fragment (16xK tile slice, K-step 32) from f16 row-major [M x ld].
// Lane layout (ISA 7.12.2, 16-bit A 16x32): lane m=l%16, g=l/16;
// halves 0..7 -> K = kb + g*8 + j ; halves 8..15 -> K = kb + 16 + g*8 + j
__device__ __forceinline__ v16h load_a_h(const _Float16* A, int M, int ld, int row0, int kb, int lane){
  int m = row0 + (lane & 15); if (m >= M) m = M - 1;
  const _Float16* p = A + (size_t)m * ld + kb + ((lane >> 4) << 3);
  v8h lo = *(const v8h*)p;
  v8h hi = *(const v8h*)(p + 16);
  v16h a;
#pragma unroll
  for (int j = 0; j < 8; ++j){ a[j] = lo[j]; a[8+j] = hi[j]; }
  return a;
}

// A fragment from f32 row-major activations, converted in-flight.
__device__ __forceinline__ v16h load_a_f(const float* A, int M, int ld, int row0, int kb, int lane){
  int m = row0 + (lane & 15); if (m >= M) m = M - 1;
  const float* p = A + (size_t)m * ld + kb + ((lane >> 4) << 3);
  v16h a;
#pragma unroll
  for (int j = 0; j < 8; ++j){ a[j] = (_Float16)p[j]; a[8+j] = (_Float16)p[16+j]; }
  return a;
}

// B fragment from transposed f16 weights Wt[Nout][K]; lane n=l%16 reads row n0+n.
__device__ __forceinline__ v16h load_b_h(const _Float16* Wt, int K, int n0, int kb, int lane){
  const _Float16* p = Wt + (size_t)(n0 + (lane & 15)) * K + kb + ((lane >> 4) << 3);
  v8h lo = *(const v8h*)p;
  v8h hi = *(const v8h*)(p + 16);
  v16h b;
#pragma unroll
  for (int j = 0; j < 8; ++j){ b[j] = lo[j]; b[8+j] = hi[j]; }
  return b;
}

// ---------------- weight prep ----------------
__global__ void conv_transpose_k(const float* __restrict__ W, _Float16* __restrict__ Wt, int In, int Out){
  int idx = blockIdx.x * blockDim.x + threadIdx.x;
  if (idx >= In * Out) return;
  int i = idx / Out, o = idx % Out;
  Wt[(size_t)o * In + i] = (_Float16)W[idx];
}

// ---------------- misc elementwise ----------------
__global__ void fill_k(float* __restrict__ p, size_t n){
  size_t i = (size_t)blockIdx.x * blockDim.x + threadIdx.x;
  if (i < n) p[i] = 0.f;
}

__global__ void embed_k(const int* __restrict__ z, const float* __restrict__ emb,
                        float* __restrict__ x, int N){
  int i = blockIdx.x * blockDim.x + threadIdx.x;
  if (i >= N * HIDC) return;
  x[i] = emb[(size_t)z[i >> 7] * HIDC + (i & 127)];
}

__global__ void concat_k(const float* __restrict__ x, const float* __restrict__ xnb,
                         float* __restrict__ xcat, int N){
  int i = blockIdx.x * blockDim.x + threadIdx.x;
  if (i >= N * 256) return;
  int n = i >> 8, c = i & 255;
  xcat[i] = (c < 128) ? x[(size_t)n * 128 + c] : xnb[(size_t)n * 128 + c - 128];
}

__global__ void vdot_k(const float* __restrict__ vp, float* __restrict__ vdot, int N){
  int i = blockIdx.x * blockDim.x + threadIdx.x;
  if (i >= N * HIDC) return;
  int n = i >> 7, c = i & 127;
  const float* base = vp + (size_t)n * 3 * 384;
  float s = 0.f;
#pragma unroll
  for (int a = 0; a < 3; ++a) s += base[a * 384 + c] * base[a * 384 + 128 + c];
  vdot[i] = s;
}

__global__ void update_k(const float* __restrict__ o, const float* __restrict__ vdot,
                         const float* __restrict__ vp, const float* __restrict__ va,
                         float* __restrict__ x, float* __restrict__ vec, int N){
  int i = blockIdx.x * blockDim.x + threadIdx.x;
  if (i >= N * HIDC) return;
  int n = i >> 7, c = i & 127;
  float o1 = o[(size_t)n * 384 + c];
  float o2 = o[(size_t)n * 384 + 128 + c];
  float o3 = o[(size_t)n * 384 + 256 + c];
  x[i] += vdot[i] * o2 + o3;
  const float* v3 = vp + (size_t)n * 3 * 384 + 256;
#pragma unroll
  for (int a = 0; a < 3; ++a){
    size_t idx = (size_t)n * 384 + a * 128 + c;
    vec[idx] += v3[a * 384 + c] * o1 + va[idx];
  }
}

__global__ void copy_batch_k(const int* __restrict__ batch, int* __restrict__ out, int N){
  int i = blockIdx.x * blockDim.x + threadIdx.x;
  if (i < N) out[i] = batch[i];
}

// ---------------- LayerNorm (one wave per 128-ch row) ----------------
__global__ void layernorm_k(const float* __restrict__ X, const float* __restrict__ g,
                            const float* __restrict__ b, float* __restrict__ Y, int N, int act){
  int lane = threadIdx.x & 31, wave = threadIdx.x >> 5;
  int row = blockIdx.x * (blockDim.x >> 5) + wave;
  if (row >= N) return;
  const float* x = X + (size_t)row * HIDC;
  float v0[4]; float s = 0.f;
#pragma unroll
  for (int j = 0; j < 4; ++j){ v0[j] = x[lane * 4 + j]; s += v0[j]; }
  for (int m = 16; m >= 1; m >>= 1) s += __shfl_xor(s, m);
  float mu = s * (1.f / HIDC);
  float var = 0.f;
#pragma unroll
  for (int j = 0; j < 4; ++j){ float d = v0[j] - mu; var += d * d; }
  for (int m = 16; m >= 1; m >>= 1) var += __shfl_xor(var, m);
  float inv = rsqrtf(var * (1.f / HIDC) + 1e-5f);
  float* y = Y + (size_t)row * HIDC;
#pragma unroll
  for (int j = 0; j < 4; ++j){
    int c = lane * 4 + j;
    float o = (v0[j] - mu) * inv * g[c] + b[c];
    if (act) o = siluf(o);
    y[c] = o;
  }
}

// ---------------- edge geometry + RBF expansion ----------------
__global__ void edge_geom_k(const float* __restrict__ pos, const int* __restrict__ esrc,
                            const int* __restrict__ edst, const float* __restrict__ means,
                            const float* __restrict__ betas, float* __restrict__ unit,
                            float* __restrict__ Cv, float* __restrict__ Cn,
                            _Float16* __restrict__ fij, int E){
  int e = blockIdx.x * blockDim.x + threadIdx.x;
  if (e >= E) return;
  int s = esrc[e], t = edst[e];
  float ex = pos[s*3+0] - pos[t*3+0];
  float ey = pos[s*3+1] - pos[t*3+1];
  float ez = pos[s*3+2] - pos[t*3+2];
  bool nonself = (s != t);
  float sq = ex*ex + ey*ey + ez*ez;
  float d  = nonself ? sqrtf(sq) : 0.f;
  float iv = (nonself && d > 0.f) ? (1.f / d) : 0.f;
  unit[e*3+0] = ex * iv; unit[e*3+1] = ey * iv; unit[e*3+2] = ez * iv;
  float C = (d < CUTV) ? 0.5f * (__cosf(d * (3.14159265358979f / CUTV)) + 1.f) : 0.f;
  Cv[e] = C;
  Cn[e] = nonself ? C : 0.f;
  float ed = __expf(-(5.f / CUTV) * d);
#pragma unroll 4
  for (int r = 0; r < RBFC; ++r){
    float df = ed - means[r];
    fij[(size_t)e * RBFC + r] = (_Float16)(C * __expf(-betas[r] * df * df));
  }
}

// ---------------- generic WMMA GEMM: out = act(A[MxK]*WtT + bias) ----------------
__global__ __launch_bounds__(256)
void gemm_a32_k(const float* __restrict__ A, const _Float16* __restrict__ Wt,
                const float* __restrict__ bias, float* __restrict__ out,
                int M, int K, int Nout, int act){
  int lane = threadIdx.x & 31, wave = threadIdx.x >> 5;
  int tiles_n = Nout >> 5;
  int tiles_m = (M + 15) >> 4;
  int tile = blockIdx.x * (blockDim.x >> 5) + wave;
  if (tile >= tiles_m * tiles_n) return;
  int tm = tile / tiles_n, tn = tile % tiles_n;
  int row0 = tm << 4, col0 = tn << 5;
  __builtin_prefetch(Wt + (size_t)col0 * K, 0, 0);
  v8f acc0 = {}; v8f acc1 = {};
#pragma unroll 2
  for (int kb = 0; kb < K; kb += 32){
    v16h a  = load_a_f(A, M, K, row0, kb, lane);
    v16h b0 = load_b_h(Wt, K, col0,      kb, lane);
    v16h b1 = load_b_h(Wt, K, col0 + 16, kb, lane);
    acc0 = wmma_f16(a, b0, acc0);
    acc1 = wmma_f16(a, b1, acc1);
  }
  int cA = col0 + (lane & 15), cB = cA + 16;
  float bA = bias ? bias[cA] : 0.f;
  float bB = bias ? bias[cB] : 0.f;
  int rbase = row0 + ((lane >> 4) << 3);
#pragma unroll
  for (int r = 0; r < 8; ++r){
    int row = rbase + r;
    if (row < M){
      float vA = acc0[r] + bA, vB = acc1[r] + bB;
      if (act){ vA = siluf(vA); vB = siluf(vB); }
      out[(size_t)row * Nout + cA] = vA;
      out[(size_t)row * Nout + cB] = vB;
    }
  }
}

// ---------------- neighbor embedding ----------------
// Weights (nb_WdT, 128x64 f16 = 16 KB) staged once per workgroup into LDS via
// the Tensor Data Mover; all 8 waves serve B fragments from LDS (ds_load),
// keeping the VMEM pipe for streaming fij A tiles + atomic scatters.
__global__ __launch_bounds__(256)
void neighbor_emb_k(const _Float16* __restrict__ fij, const _Float16* __restrict__ WdT,
                    const float* __restrict__ bd, const float* __restrict__ Cn,
                    const float* __restrict__ nb_emb, const int* __restrict__ z,
                    const int* __restrict__ esrc, const int* __restrict__ edst,
                    float* __restrict__ xnb, int E){
  __shared__ _Float16 wlds[HIDC * RBFC];   // 16 KB, LDS offset 0
  int lane = threadIdx.x & 31, wave = threadIdx.x >> 5;
  int tile = blockIdx.x * (blockDim.x >> 5) + wave;
  bool active = tile < ((E + 15) >> 4);

  if (wave == 0){
    // 1-D TDM descriptor: 2048 x 8B contiguous -> LDS 0
    unsigned long long ga = (unsigned long long)WdT;
    u32x4 g0 = { 1u,                                   // count = 1
                 0u,                                   // lds_addr = 0
                 (unsigned)(ga & 0xFFFFFFFFu),
                 (unsigned)((ga >> 32) & 0x1FFFFFFu) | (2u << 30) };  // type=2
    i32x8 g1 = { (3 << 16),              // data_size = 8B
                 (int)(2048u << 16),     // tensor_dim0[15:0] (bits 63:48)
                 (1 << 16),              // tensor_dim1 = 1
                 (int)(2048u << 16),     // tile_dim0 = 2048 (bits 127:112)
                 0,                      // tile_dim1/2 unused
                 2048,                   // tensor_dim0_stride lo32
                 0, 0 };
    i32x4 gz = { 0, 0, 0, 0 };
#if __clang_major__ >= 23
    i32x8 gz8 = { 0, 0, 0, 0, 0, 0, 0, 0 };
    __builtin_amdgcn_tensor_load_to_lds(g0, g1, gz, gz, gz8, 0);
#else
    __builtin_amdgcn_tensor_load_to_lds(g0, g1, gz, gz, 0);
#endif
    __builtin_amdgcn_s_wait_tensorcnt(0);
  }
  __syncthreads();
  if (!active) return;

  auto ldb = [&](int n0, int kb) -> v16h {
    const _Float16* p = &wlds[((n0 + (lane & 15)) << 6) + kb + ((lane >> 4) << 3)];
    v8h lo = *(const v8h*)p;
    v8h hi = *(const v8h*)(p + 16);
    v16h b;
#pragma unroll
    for (int j = 0; j < 8; ++j){ b[j] = lo[j]; b[8+j] = hi[j]; }
    return b;
  };

  int e0 = tile << 4;
  v16h a0 = load_a_h(fij, E, RBFC, e0, 0,  lane);
  v16h a1 = load_a_h(fij, E, RBFC, e0, 32, lane);
  int rbase = (lane >> 4) << 3;
#pragma unroll
  for (int nt = 0; nt < HIDC / 16; ++nt){
    v8f acc = {};
    acc = wmma_f16(a0, ldb(nt * 16, 0),  acc);
    acc = wmma_f16(a1, ldb(nt * 16, 32), acc);
    int col = nt * 16 + (lane & 15);
    float bb = bd[col];
#pragma unroll
    for (int r = 0; r < 8; ++r){
      int e = e0 + rbase + r;
      if (e < E){
        float wv = (acc[r] + bb) * Cn[e];
        if (wv != 0.f){
          float msg = nb_emb[(size_t)z[esrc[e]] * HIDC + col] * wv;
          atomicAdd(&xnb[(size_t)edst[e] * HIDC + col], msg);
        }
      }
    }
  }
}

// ---------------- fused dk/dv GEMM + attention + message scatter ----------------
__global__ __launch_bounds__(32)
void attn_edge_k(const _Float16* __restrict__ fij,
                 const _Float16* __restrict__ WdkT, const float* __restrict__ bdk,
                 const _Float16* __restrict__ WdvT, const float* __restrict__ bdv,
                 const float* __restrict__ q, const float* __restrict__ k,
                 const float* __restrict__ v, const float* __restrict__ vec,
                 const float* __restrict__ unit, const float* __restrict__ Cv,
                 const int* __restrict__ esrc, const int* __restrict__ edst,
                 float* __restrict__ xa, float* __restrict__ va, int E){
  __shared__ float lds[16 * 512];   // [edge][0:128 dk | 128:512 dv]
  int lane = threadIdx.x;
  int tile = blockIdx.x;
  if (tile >= ((E + 15) >> 4)) return;
  int e0 = tile << 4;
  __builtin_prefetch(WdkT, 0, 0);
  __builtin_prefetch(WdvT, 0, 0);
  v16h a0 = load_a_h(fij, E, RBFC, e0, 0,  lane);
  v16h a1 = load_a_h(fij, E, RBFC, e0, 32, lane);
  int rbase = (lane >> 4) << 3;
  // dk: 8 column tiles
#pragma unroll
  for (int nt = 0; nt < 8; ++nt){
    v8f acc = {};
    acc = wmma_f16(a0, load_b_h(WdkT, RBFC, nt * 16, 0,  lane), acc);
    acc = wmma_f16(a1, load_b_h(WdkT, RBFC, nt * 16, 32, lane), acc);
    int col = nt * 16 + (lane & 15);
    float bb = bdk[col];
#pragma unroll
    for (int r = 0; r < 8; ++r) lds[(rbase + r) * 512 + col] = siluf(acc[r] + bb);
  }
  // dv: 24 column tiles
#pragma unroll 4
  for (int nt = 0; nt < 24; ++nt){
    v8f acc = {};
    acc = wmma_f16(a0, load_b_h(WdvT, RBFC, nt * 16, 0,  lane), acc);
    acc = wmma_f16(a1, load_b_h(WdvT, RBFC, nt * 16, 32, lane), acc);
    int col = nt * 16 + (lane & 15);
    float bb = bdv[col];
#pragma unroll
    for (int r = 0; r < 8; ++r) lds[(rbase + r) * 512 + 128 + col] = siluf(acc[r] + bb);
  }
  __syncthreads();
  int c0 = lane * 4;           // 4 channels per lane, head = lane/4 (16 ch/head)
  for (int e = 0; e < 16; ++e){
    int edge = e0 + e;
    if (edge >= E) break;      // uniform across wave
    int s = esrc[edge], t = edst[edge];
    const float* qs = q + (size_t)t * HIDC;
    const float* ks = k + (size_t)s * HIDC;
    float part = 0.f;
#pragma unroll
    for (int j = 0; j < 4; ++j){
      int c = c0 + j;
      part += qs[c] * ks[c] * lds[e * 512 + c];
    }
    part += __shfl_xor(part, 1);
    part += __shfl_xor(part, 2);           // per-head sum within lane-group of 4
    float attn = siluf(part) * Cv[edge];
    float ux = unit[edge*3+0], uy = unit[edge*3+1], uz = unit[edge*3+2];
    const float* vs   = v   + (size_t)s * 384;
    const float* vecs = vec + (size_t)s * 384;
#pragma unroll
    for (int j = 0; j < 4; ++j){
      int c = c0 + j;
      float vm = vs[c]         * lds[e * 512 + 128 + c];
      float w1 = vs[128 + c]   * lds[e * 512 + 256 + c];
      float w2 = vs[256 + c]   * lds[e * 512 + 384 + c];
      atomicAdd(&xa[(size_t)t * HIDC + c], vm * attn);
      atomicAdd(&va[(size_t)t * 384 + c],        vecs[c]       * w1 + w2 * ux);
      atomicAdd(&va[(size_t)t * 384 + 128 + c],  vecs[128 + c] * w1 + w2 * uy);
      atomicAdd(&va[(size_t)t * 384 + 256 + c],  vecs[256 + c] * w1 + w2 * uz);
    }
  }
}

// =====================================================================
extern "C" void kernel_launch(void* const* d_in, const int* in_sizes, int n_in,
                              void* d_out, int out_size, void* d_ws, size_t ws_size,
                              hipStream_t stream) {
  (void)n_in; (void)out_size; (void)ws_size;
  const int N = in_sizes[0];
  const int E = in_sizes[3];

  const int*   z      = (const int*)d_in[0];
  const float* pos    = (const float*)d_in[1];
  const int*   batch  = (const int*)d_in[2];
  const int*   esrc   = (const int*)d_in[3];
  const int*   edst   = (const int*)d_in[4];
  const float* emb    = (const float*)d_in[5];
  const float* nb_emb = (const float*)d_in[6];
  const float* nb_Wd  = (const float*)d_in[7];
  const float* nb_bd  = (const float*)d_in[8];
  const float* nb_Wc  = (const float*)d_in[9];
  const float* nb_bc  = (const float*)d_in[10];
  const float* means  = (const float*)d_in[11];
  const float* betas  = (const float*)d_in[12];
  const float* ln_g   = (const float*)d_in[13];
  const float* ln_b   = (const float*)d_in[14];
  const float* Wq     = (const float*)d_in[15];
  const float* bq     = (const float*)d_in[16];
  const float* Wk     = (const float*)d_in[17];
  const float* bk     = (const float*)d_in[18];
  const float* Wv     = (const float*)d_in[19];
  const float* bv     = (const float*)d_in[20];
  const float* Wo     = (const float*)d_in[21];
  const float* bo     = (const float*)d_in[22];
  const float* Wvec   = (const float*)d_in[23];
  const float* Wdk    = (const float*)d_in[24];
  const float* bdk    = (const float*)d_in[25];
  const float* Wdv    = (const float*)d_in[26];
  const float* bdv    = (const float*)d_in[27];
  const float* og     = (const float*)d_in[28];
  const float* ob     = (const float*)d_in[29];
  const float* Wp     = (const float*)d_in[30];
  const float* bp     = (const float*)d_in[31];
  const float* pg     = (const float*)d_in[32];
  const float* pb     = (const float*)d_in[33];

  size_t off = 0;
  auto alloc = [&](size_t bytes) -> void* {
    void* r = (char*)d_ws + off;
    off += (bytes + 255) & ~(size_t)255;
    return r;
  };

  // f16 transposed weights
  _Float16* nbWdT = (_Float16*)alloc(128 * 64 * 2);
  _Float16* nbWcT = (_Float16*)alloc(128 * 256 * 2);
  _Float16* WqT   = (_Float16*)alloc((size_t)3 * 128 * 128 * 2);
  _Float16* WkT   = (_Float16*)alloc((size_t)3 * 128 * 128 * 2);
  _Float16* WvT   = (_Float16*)alloc((size_t)3 * 384 * 128 * 2);
  _Float16* WoT   = (_Float16*)alloc((size_t)3 * 384 * 128 * 2);
  _Float16* WvecT = (_Float16*)alloc((size_t)3 * 384 * 128 * 2);
  _Float16* WdkT  = (_Float16*)alloc((size_t)3 * 128 * 64 * 2);
  _Float16* WdvT  = (_Float16*)alloc((size_t)3 * 384 * 64 * 2);
  _Float16* WpT   = (_Float16*)alloc(128 * 128 * 2);
  // edge buffers
  _Float16* fijh  = (_Float16*)alloc((size_t)E * RBFC * 2);
  float* unit = (float*)alloc((size_t)E * 3 * 4);
  float* Cvb  = (float*)alloc((size_t)E * 4);
  float* Cnb  = (float*)alloc((size_t)E * 4);
  // node buffers
  float* x    = (float*)alloc((size_t)N * 128 * 4);
  float* xnb  = (float*)alloc((size_t)N * 128 * 4);
  float* xcat = (float*)alloc((size_t)N * 256 * 4);
  float* xn   = (float*)alloc((size_t)N * 128 * 4);
  float* qb   = (float*)alloc((size_t)N * 128 * 4);
  float* kb   = (float*)alloc((size_t)N * 128 * 4);
  float* vb   = (float*)alloc((size_t)N * 384 * 4);
  float* vec  = (float*)alloc((size_t)N * 384 * 4);
  float* vp   = (float*)alloc((size_t)N * 1152 * 4);
  float* vdot = (float*)alloc((size_t)N * 128 * 4);
  float* xa   = (float*)alloc((size_t)N * 128 * 4);
  float* va   = (float*)alloc((size_t)N * 384 * 4);
  float* ob_  = (float*)alloc((size_t)N * 384 * 4);
  float* tmp  = (float*)alloc((size_t)N * 128 * 4);

  auto ct = [&](const float* W, _Float16* Wt, int In, int Out){
    int n = In * Out;
    conv_transpose_k<<<(n + 255) / 256, 256, 0, stream>>>(W, Wt, In, Out);
  };
  ct(nb_Wd, nbWdT, 64, 128);
  ct(nb_Wc, nbWcT, 256, 128);
  ct(Wp, WpT, 128, 128);
  for (int l = 0; l < NLAY; ++l){
    ct(Wq   + (size_t)l * 128 * 128, WqT   + (size_t)l * 128 * 128, 128, 128);
    ct(Wk   + (size_t)l * 128 * 128, WkT   + (size_t)l * 128 * 128, 128, 128);
    ct(Wv   + (size_t)l * 128 * 384, WvT   + (size_t)l * 384 * 128, 128, 384);
    ct(Wo   + (size_t)l * 128 * 384, WoT   + (size_t)l * 384 * 128, 128, 384);
    ct(Wvec + (size_t)l * 128 * 384, WvecT + (size_t)l * 384 * 128, 128, 384);
    ct(Wdk  + (size_t)l * 64 * 128,  WdkT  + (size_t)l * 128 * 64,  64, 128);
    ct(Wdv  + (size_t)l * 64 * 384,  WdvT  + (size_t)l * 384 * 64,  64, 384);
  }

  auto zero = [&](float* p, size_t n){
    fill_k<<<(unsigned)((n + 255) / 256), 256, 0, stream>>>(p, n);
  };
  auto gemm = [&](const float* A, const _Float16* Wt, const float* bias, float* out_,
                  int M, int K, int Nout){
    int tiles = ((M + 15) >> 4) * (Nout >> 5);
    gemm_a32_k<<<(tiles + 7) / 8, 256, 0, stream>>>(A, Wt, bias, out_, M, K, Nout, 0);
  };

  int etiles = (E + 15) >> 4;

  // --- embedding + edge geometry + neighbor embedding ---
  embed_k<<<(N * 128 + 255) / 256, 256, 0, stream>>>(z, emb, x, N);
  zero(xnb, (size_t)N * 128);
  zero(vec, (size_t)N * 384);
  edge_geom_k<<<(E + 255) / 256, 256, 0, stream>>>(pos, esrc, edst, means, betas,
                                                   unit, Cvb, Cnb, fijh, E);
  neighbor_emb_k<<<(etiles + 7) / 8, 256, 0, stream>>>(fijh, nbWdT, nb_bd, Cnb,
                                                       nb_emb, z, esrc, edst, xnb, E);
  concat_k<<<(N * 256 + 255) / 256, 256, 0, stream>>>(x, xnb, xcat, N);
  gemm(xcat, nbWcT, nb_bc, x, N, 256, 128);

  // --- transformer layers ---
  for (int l = 0; l < NLAY; ++l){
    layernorm_k<<<(N + 7) / 8, 256, 0, stream>>>(x, ln_g + l * 128, ln_b + l * 128, xn, N, 0);
    gemm(xn, WqT + (size_t)l * 128 * 128, bq + l * 128, qb, N, 128, 128);
    gemm(xn, WkT + (size_t)l * 128 * 128, bk + l * 128, kb, N, 128, 128);
    gemm(xn, WvT + (size_t)l * 384 * 128, bv + l * 384, vb, N, 128, 384);
    gemm(vec, WvecT + (size_t)l * 384 * 128, nullptr, vp, 3 * N, 128, 384);
    vdot_k<<<(N * 128 + 255) / 256, 256, 0, stream>>>(vp, vdot, N);
    zero(xa, (size_t)N * 128);
    zero(va, (size_t)N * 384);
    attn_edge_k<<<etiles, 32, 0, stream>>>(fijh,
        WdkT + (size_t)l * 128 * 64, bdk + l * 128,
        WdvT + (size_t)l * 384 * 64, bdv + l * 384,
        qb, kb, vb, vec, unit, Cvb, esrc, edst, xa, va, E);
    gemm(xa, WoT + (size_t)l * 384 * 128, bo + l * 384, ob_, N, 128, 384);
    update_k<<<(N * 128 + 255) / 256, 256, 0, stream>>>(ob_, vdot, vp, va, x, vec, N);
  }

  // --- output head ---
  layernorm_k<<<(N + 7) / 8, 256, 0, stream>>>(x, og, ob, xn, N, 0);
  gemm(xn, WpT, bp, tmp, N, 128, 128);
  layernorm_k<<<(N + 7) / 8, 256, 0, stream>>>(tmp, pg, pb, (float*)d_out, N, 1);
  copy_batch_k<<<(N + 255) / 256, 256, 0, stream>>>(batch, (int*)((float*)d_out + (size_t)N * 128), N);
}
